// AttnDecoder_3341484556812
// MI455X (gfx1250) — compile-verified, matrix operand order checked by reading, not run on hardware
//
#include <hip/hip_runtime.h>
#include <hip/hip_bf16.h>

typedef __bf16 bf16t;
typedef __attribute__((ext_vector_type(16))) __bf16 v16bf;
typedef __attribute__((ext_vector_type(8)))  __bf16 v8bf;
typedef __attribute__((ext_vector_type(8)))  float  v8f;

static constexpr int B_  = 50;
static constexpr int T_  = 200;
static constexpr int S_  = 40;
static constexpr int H_  = 8;
static constexpr int DO_ = 256;
static constexpr int DZ_ = 512;
static constexpr int KLD_OFF = B_ * (T_ + 1) * DO_;   // 2,572,800 floats

// ---------------------------------------------------------------------------
// fp32 -> bf16 (row-major) conversion
// ---------------------------------------------------------------------------
__global__ void f2bf_kernel(const float* __restrict__ src, bf16t* __restrict__ dst, int n) {
    int i = blockIdx.x * blockDim.x + threadIdx.x;
    if (i < n) dst[i] = (__bf16)src[i];
}

// ---------------------------------------------------------------------------
// Pack fp32 weight W[K,N] into bf16 WMMA-B-fragment order:
//   Wp[(kb*ntn + tn)*32 + lane][16]  (16x16x32 bf16 B lane layout:
//   lane = n + 16*half; half=0 -> K-octets {0..7,16..23}, half=1 -> {8..15,24..31})
// ---------------------------------------------------------------------------
__global__ void pack_w_kernel(const float* __restrict__ W, bf16t* __restrict__ Wp,
                              int N, int K) {
    int ntn = N >> 4;
    int total = (K >> 5) * ntn * 32;
    int tix = blockIdx.x * blockDim.x + threadIdx.x;
    if (tix >= total) return;
    int lane = tix & 31;
    int tmp = tix >> 5;
    int tn = tmp % ntn;
    int kb = tmp / ntn;
    int half = lane >> 4, idx = lane & 15;
    int bcol = tn * 16 + idx;
    int k0 = kb * 32;
    bf16t* dst = Wp + (size_t)tix * 16;
#pragma unroll
    for (int j = 0; j < 16; ++j) {
        int kk = k0 + j + ((j < 8) ? (half << 3) : 8 + (half << 3));
        dst[j] = (__bf16)W[(size_t)kk * N + bcol];
    }
}

// ---------------------------------------------------------------------------
// bf16 WMMA GEMM: C[M,N] = act( A[M,K](bf16 row-major) * W(packed) + bias ).
// One wave computes a 16x64 strip (4 N-tiles share one A fragment).
// A rows clamped: out-of-range rows feed only C rows >= M, never stored.
// A fragment = two b128 loads (no cvt in the loop); each B tile = one 32B load.
// ---------------------------------------------------------------------------
template <bool OBF>
__global__ void gemm_bf16_kernel(const bf16t* __restrict__ A, int lda,
                                 const bf16t* __restrict__ Wp,
                                 const float* __restrict__ bias,
                                 void* __restrict__ Cv, int ldc,
                                 int M, int N, int K, int act) {
    int wave = threadIdx.x >> 5;
    int lane = threadIdx.x & 31;
    int ntn = N >> 4;
    int ngrp = ntn >> 2;                 // groups of 4 N-tiles
    int ntm = (M + 15) >> 4;
    int tile = blockIdx.x * 4 + wave;
    if (tile >= ntm * ngrp) return;
    int tm = tile / ngrp, tn0 = (tile % ngrp) << 2;
    int half = lane >> 4, idx = lane & 15;
    int arow = tm * 16 + idx;
    if (arow >= M) arow = M - 1;
    const bf16t* Ar = A + (size_t)arow * lda + (half << 3);
    const bf16t* Bp = Wp + ((size_t)tn0 * 32 + lane) * 16;
    const size_t bstride = (size_t)ntn * 512;  // packed elems per k-block

    v8f acc0 = {0.f,0.f,0.f,0.f,0.f,0.f,0.f,0.f};
    v8f acc1 = acc0, acc2 = acc0, acc3 = acc0;

#pragma unroll 2
    for (int k0 = 0; k0 < K; k0 += 32) {
        v8bf al = *(const v8bf*)(Ar + k0);
        v8bf ah = *(const v8bf*)(Ar + k0 + 16);
        v16bf b0 = *(const v16bf*)(Bp);
        v16bf b1 = *(const v16bf*)(Bp + 512);
        v16bf b2 = *(const v16bf*)(Bp + 1024);
        v16bf b3 = *(const v16bf*)(Bp + 1536);
        Bp += bstride;
        if (k0 + 32 < K) __builtin_prefetch(Bp, 0, 1);  // global_prefetch_b8
        v16bf a = __builtin_shufflevector(al, ah,
            0,1,2,3,4,5,6,7,8,9,10,11,12,13,14,15);
        acc0 = __builtin_amdgcn_wmma_f32_16x16x32_bf16(false, a, false, b0, (short)0, acc0, false, false);
        acc1 = __builtin_amdgcn_wmma_f32_16x16x32_bf16(false, a, false, b1, (short)0, acc1, false, false);
        acc2 = __builtin_amdgcn_wmma_f32_16x16x32_bf16(false, a, false, b2, (short)0, acc2, false, false);
        acc3 = __builtin_amdgcn_wmma_f32_16x16x32_bf16(false, a, false, b3, (short)0, acc3, false, false);
    }

#pragma unroll
    for (int u = 0; u < 4; ++u) {
        v8f acc = (u == 0) ? acc0 : (u == 1) ? acc1 : (u == 2) ? acc2 : acc3;
        int col = (tn0 + u) * 16 + idx;
        float bi = bias ? bias[col] : 0.f;
#pragma unroll
        for (int r = 0; r < 8; ++r) {
            int row = tm * 16 + r + (half << 3);
            if (row >= M) continue;
            float v = acc[r] + bi;
            if (act == 1)      v = v > 0.f ? v : 0.f;
            else if (act == 2) v = v >= 0.f ? v : 0.2f * v;
            if (OBF) ((bf16t*)Cv)[(size_t)row * ldc + col] = (__bf16)v;
            else     ((float*)Cv)[(size_t)row * ldc + col] = v;
        }
    }
}

// ---------------------------------------------------------------------------
// LayerNorm over 512 cols (+optional residual, optional lrelu).
// Writes fp32 (residual path) and/or bf16 (GEMM-A path) outputs.
// ---------------------------------------------------------------------------
__global__ void ln512_kernel(const float* __restrict__ xin,
                             const float* __restrict__ res,
                             const float* __restrict__ g,
                             const float* __restrict__ b,
                             float* __restrict__ outf,
                             bf16t* __restrict__ outh, int act) {
    __shared__ float sm[512];
    int row = blockIdx.x, i = threadIdx.x;
    float v = xin[row * 512 + i] + (res ? res[row * 512 + i] : 0.f);
    sm[i] = v;
    __syncthreads();
    for (int s = 256; s > 0; s >>= 1) { if (i < s) sm[i] += sm[i + s]; __syncthreads(); }
    float mean = sm[0] * (1.0f / 512.0f);
    __syncthreads();
    float d = v - mean;
    sm[i] = d * d;
    __syncthreads();
    for (int s = 256; s > 0; s >>= 1) { if (i < s) sm[i] += sm[i + s]; __syncthreads(); }
    float var = sm[0] * (1.0f / 512.0f);
    float y = d * rsqrtf(var + 1e-5f) * g[i] + b[i];
    if (act == 2) y = y >= 0.f ? y : 0.2f * y;
    if (outf) outf[row * 512 + i] = y;
    if (outh) outh[row * 512 + i] = (__bf16)y;
}

// ---------------------------------------------------------------------------
// Single-query attention over nkeys keys (bf16 K/V caches), per (b,h) block.
// q staged in LDS; K-row dot vectorized (b128 bf16 loads). Output bf16.
// ---------------------------------------------------------------------------
__global__ void attn_kernel(const float* __restrict__ q,
                            const bf16t* __restrict__ kmat,
                            const bf16t* __restrict__ vmat,
                            bf16t* __restrict__ out,
                            int nkeys, int batch_stride) {
    __shared__ float sc[256];
    __shared__ float red[256];
    __shared__ float qs[64];
    int b = blockIdx.x >> 3;
    int h = blockIdx.x & 7;
    int tid = threadIdx.x;
    const bf16t* kb = kmat + (size_t)b * batch_stride + h * 64;
    const bf16t* vb = vmat + (size_t)b * batch_stride + h * 64;

    if (tid < 64) qs[tid] = q[b * 512 + h * 64 + tid];
    __syncthreads();

    float s = -1e30f;
    if (tid < nkeys) {
        const v8bf* kr = (const v8bf*)(kb + (size_t)tid * 512);
        float acc = 0.f;
#pragma unroll
        for (int c = 0; c < 8; ++c) {
            v8bf kv = kr[c];
#pragma unroll
            for (int j = 0; j < 8; ++j) acc += qs[c * 8 + j] * (float)kv[j];
        }
        s = acc * 0.125f;  // 1/sqrt(64)
    }
    sc[tid] = s; red[tid] = s;
    __syncthreads();
    for (int st = 128; st > 0; st >>= 1) { if (tid < st) red[tid] = fmaxf(red[tid], red[tid + st]); __syncthreads(); }
    float mx = red[0];
    __syncthreads();
    float e = (tid < nkeys) ? __expf(s - mx) : 0.f;
    sc[tid] = e; red[tid] = e;
    __syncthreads();
    for (int st = 128; st > 0; st >>= 1) { if (tid < st) red[tid] += red[tid + st]; __syncthreads(); }
    float inv = 1.f / red[0];
    if (tid < 64) {
        float o = 0.f;
        for (int j = 0; j < nkeys; ++j) o += sc[j] * (float)vb[(size_t)j * 512 + tid];
        out[b * 512 + h * 64 + tid] = (__bf16)(o * inv);
    }
}

// ---------------------------------------------------------------------------
// Step-input builders & pointwise kernels
// ---------------------------------------------------------------------------
__global__ void build_x_kernel(const float* __restrict__ prev, float* __restrict__ x,
                               bf16t* __restrict__ xh, int t) {
    int i = blockIdx.x * blockDim.x + threadIdx.x;
    if (i >= B_ * 512) return;
    int b = i >> 9, d = i & 511;
    float v;
    if (d < 256) v = prev[b * 256 + d];
    else {
        int j = d - 256;
        float angle = (float)t * __expf(-(float)(2 * (j / 2)) * (1.0f / 256.0f) * 9.210340372f);
        v = (j & 1) ? __cosf(angle) : __sinf(angle);
    }
    x[i] = v;
    xh[i] = (__bf16)v;
}

__global__ void build_headin_kernel(const float* __restrict__ x0, const float* __restrict__ att,
                                    const float* __restrict__ inputemb,
                                    bf16t* __restrict__ pos_in, bf16t* __restrict__ pri_in, int t) {
    int i = blockIdx.x * blockDim.x + threadIdx.x;
    if (i >= B_ * 1280) return;
    int b = i / 1280, c = i % 1280;
    float v;
    if (c < 512) v = x0[b * 512 + c];
    else if (c < 768) {
        float m = inputemb[((size_t)b * T_ + t) * 256 + (c - 512)];
        v = (m == __builtin_inff()) ? 0.f : m;
    } else v = att[b * 512 + (c - 768)];
    pos_in[i] = (__bf16)v;
    if (c < 512)       pri_in[b * 1024 + c] = (__bf16)v;
    else if (c >= 768) pri_in[b * 1024 + 512 + (c - 768)] = (__bf16)v;
}

__global__ void build_decin_kernel(const float* __restrict__ x0, const float* __restrict__ att,
                                   const float* __restrict__ z, bf16t* __restrict__ dec_in) {
    int i = blockIdx.x * blockDim.x + threadIdx.x;
    if (i >= B_ * 1536) return;
    int b = i / 1536, c = i % 1536;
    float v;
    if (c < 512)       v = x0[b * 512 + c];
    else if (c < 1024) v = att[b * 512 + (c - 512)];
    else               v = z[b * 512 + (c - 1024)];
    dec_in[i] = (__bf16)v;
}

__device__ __forceinline__ unsigned hmix(unsigned h) {
    h ^= h >> 16; h *= 0x7FEB352Du; h ^= h >> 15; h *= 0x846CA68Bu; h ^= h >> 16;
    return h;
}
__device__ __forceinline__ float nrand(unsigned strm, unsigned t, unsigned i) {
    unsigned u1 = hmix(0x9E3779B1u * (strm * 1315423911u + t) ^ (i * 0x85EBCA77u) ^ 0xA511E9B3u);
    unsigned u2 = hmix(0xC2B2AE3Du * (strm * 2654435761u + t) ^ (i * 0x27D4EB2Fu) ^ 0x165667B1u);
    float f1 = ((float)u1 + 1.0f) * (1.0f / 4294967296.0f);
    float f2 = (float)u2 * (1.0f / 4294967296.0f);
    return sqrtf(-2.0f * logf(f1)) * __cosf(6.28318530718f * f2);
}

__global__ void z_kernel(const float* __restrict__ mu, const float* __restrict__ lv,
                         float* __restrict__ z, int t) {
    int i = blockIdx.x * blockDim.x + threadIdx.x;
    if (i >= B_ * DZ_) return;
    float eps = nrand(0u, (unsigned)t, (unsigned)i);
    z[i] = eps * __expf(0.5f * lv[i]) + mu[i];
}

__global__ void kld_kernel(const float* __restrict__ mu1, const float* __restrict__ lv1,
                           const float* __restrict__ mu2, const float* __restrict__ lv2,
                           float* __restrict__ out, int t) {
    int i = blockIdx.x * blockDim.x + threadIdx.x;
    if (i >= B_ * DZ_) return;
    int b = i >> 9, d = i & 511;
    float m1 = mu1[i], l1 = lv1[i], m2 = mu2[i], l2 = lv2[i];
    float dm = m1 - m2;
    float v = 0.5f * (l2 - l1) + (__expf(l1) + dm * dm) / (2.f * __expf(l2)) - 0.5f;
    out[((size_t)b * T_ + t) * DZ_ + d] = v;
}

__global__ void fake_kernel(const float* __restrict__ dout, float* __restrict__ trg_out,
                            float* __restrict__ prev, int t) {
    int i = blockIdx.x * blockDim.x + threadIdx.x;
    if (i >= B_ * DO_) return;
    int b = i >> 8, d = i & 255;
    float v = dout[i];
    if (d == 255) v = 1.f / (1.f + __expf(-v));
    prev[i] = v;
    trg_out[((size_t)b * (T_ + 1) + (t + 1)) * DO_ + d] = v;
}

__global__ void init_kernel(const float* __restrict__ trg_seq, float* __restrict__ trg_out,
                            float* __restrict__ prev) {
    int i = blockIdx.x * blockDim.x + threadIdx.x;
    if (i >= B_ * DO_) return;
    int b = i >> 8, d = i & 255;
    float v = trg_seq[(size_t)b * (T_ + 1) * DO_ + d];
    trg_out[(size_t)b * (T_ + 1) * DO_ + d] = v;
    prev[i] = v;
}

// ---------------------------------------------------------------------------
// Host side
// ---------------------------------------------------------------------------
static void launch_gemm(hipStream_t st, const bf16t* A, int lda, const bf16t* Wp,
                        const float* bias, void* C, int ldc,
                        int M, int N, int K, int act, bool obf) {
    int tiles = ((M + 15) / 16) * (N / 64);   // 4 N-tiles per wave
    int blocks = (tiles + 3) / 4;
    if (obf) gemm_bf16_kernel<true ><<<blocks, 128, 0, st>>>(A, lda, Wp, bias, C, ldc, M, N, K, act);
    else     gemm_bf16_kernel<false><<<blocks, 128, 0, st>>>(A, lda, Wp, bias, C, ldc, M, N, K, act);
}

static void launch_pack(hipStream_t st, const float* W, bf16t* Wp, int K, int N) {
    int total = (K >> 5) * (N >> 4) * 32;
    pack_w_kernel<<<(total + 255) / 256, 256, 0, st>>>(W, Wp, N, K);
}

extern "C" void kernel_launch(void* const* d_in, const int* in_sizes, int n_in,
                              void* d_out, int out_size, void* d_ws, size_t ws_size,
                              hipStream_t stream) {
    (void)in_sizes; (void)n_in; (void)out_size; (void)ws_size;
    const float* inputemb = (const float*)d_in[0];
    const float* trg_seq  = (const float*)d_in[1];
    const float* enc_out  = (const float*)d_in[3];
    auto P = [&](int i) { return (const float*)d_in[i]; };

    // params flatten order (jax pytree: dict keys sorted):
    //   layers[0..1] (20 leaves each), seq_dec (10), seq_post (10), seq_pri (10)
    const int LB[2] = {5, 25};
    const int DEC = 45; // W1,W2,We,b1,b2,bb1,bbe,be,g1,ge
    const int POS = 55; // W1,We,Wlv,Wmu,b1,bbe,be,blv,bmu,ge
    const int PRI = 65;

    char* ws = (char*)d_ws;
    size_t off = 0;
    auto alloc = [&](size_t bytes) -> char* {
        off = (off + 255) & ~(size_t)255;
        char* p = ws + off; off += bytes; return p;
    };
    auto ab = [&](size_t elems) { return (bf16t*)alloc(elems * sizeof(bf16t)); };
    auto af = [&](size_t elems) { return (float*)alloc(elems * sizeof(float)); };

    // packed bf16 weights
    bf16t *wq[2], *wk[2], *wv[2], *wo[2], *wq2[2], *wk2[2], *wv2[2], *wo2[2], *wf1[2], *wf2[2];
    for (int l = 0; l < 2; ++l) {
        wq[l]  = ab(262144); wk[l]  = ab(262144); wv[l]  = ab(262144); wo[l]  = ab(262144);
        wq2[l] = ab(262144); wk2[l] = ab(262144); wv2[l] = ab(262144); wo2[l] = ab(262144);
        wf1[l] = ab(1048576); wf2[l] = ab(1048576);
    }
    bf16t* pWe = ab(1280 * 512); bf16t* pW1 = ab(262144); bf16t* pWmu = ab(262144); bf16t* pWlv = ab(262144);
    bf16t* rWe = ab(1024 * 512); bf16t* rW1 = ab(262144); bf16t* rWmu = ab(262144); bf16t* rWlv = ab(262144);
    bf16t* dWe = ab(1536 * 512); bf16t* dW1 = ab(262144); bf16t* dW2 = ab(131072);
    // bf16 KV caches (row-major, attn-kernel operands)
    bf16t *kc[2], *vc[2], *ke[2], *ve[2];
    for (int l = 0; l < 2; ++l) { kc[l] = ab((size_t)B_ * T_ * 512); vc[l] = ab((size_t)B_ * T_ * 512); }
    for (int l = 0; l < 2; ++l) { ke[l] = ab((size_t)B_ * S_ * 512); ve[l] = ab((size_t)B_ * S_ * 512); }
    // bf16 activation mirrors (GEMM-A operands)
    bf16t* ench   = ab((size_t)B_ * S_ * 512);
    bf16t* x0h    = ab(B_ * 512);  bf16t* xbh   = ab(B_ * 512);
    bf16t* hbh    = ab(B_ * 512);  bf16t* h2bh  = ab(B_ * 512);
    bf16t* aobh   = ab(B_ * 512);  bf16t* hph   = ab(B_ * 512);
    bf16t* hd2h   = ab(B_ * 512);  bf16t* ffbh  = ab(B_ * 2048);
    bf16t* hidbh  = ab(B_ * 512);
    bf16t* posinh = ab(B_ * 1280); bf16t* priinh = ab(B_ * 1024); bf16t* decinh = ab(B_ * 1536);
    // fp32 buffers
    float* x0   = af(B_ * 512);  float* xb   = af(B_ * 512);
    float* qb   = af(B_ * 512);  float* sab  = af(B_ * 512);
    float* hb   = af(B_ * 512);  float* h2b  = af(B_ * 512);
    float* tmp1 = af(B_ * 512);
    float* mu1b = af(B_ * 512);  float* lv1b = af(B_ * 512);
    float* mu2b = af(B_ * 512);  float* lv2b = af(B_ * 512);
    float* zb   = af(B_ * 512);  float* doutb = af(B_ * 256);
    float* prev = af(B_ * 256);

    // --- weight packing + encoder activation conversion (once per call) ---
    struct Pk { int idx; bf16t* dst; int K, N; };
    Pk pk[31]; int np = 0;
    for (int l = 0; l < 2; ++l) {
        pk[np++] = {LB[l] + 6, wq[l], 512, 512};  pk[np++] = {LB[l] + 2, wk[l], 512, 512};
        pk[np++] = {LB[l] + 8, wv[l], 512, 512};  pk[np++] = {LB[l] + 4, wo[l], 512, 512};
        pk[np++] = {LB[l] + 7, wq2[l], 512, 512}; pk[np++] = {LB[l] + 3, wk2[l], 512, 512};
        pk[np++] = {LB[l] + 9, wv2[l], 512, 512}; pk[np++] = {LB[l] + 5, wo2[l], 512, 512};
        pk[np++] = {LB[l] + 0, wf1[l], 512, 2048}; pk[np++] = {LB[l] + 1, wf2[l], 2048, 512};
    }
    pk[np++] = {POS + 1, pWe, 1280, 512}; pk[np++] = {POS + 0, pW1, 512, 512};
    pk[np++] = {POS + 3, pWmu, 512, 512}; pk[np++] = {POS + 2, pWlv, 512, 512};
    pk[np++] = {PRI + 1, rWe, 1024, 512}; pk[np++] = {PRI + 0, rW1, 512, 512};
    pk[np++] = {PRI + 3, rWmu, 512, 512}; pk[np++] = {PRI + 2, rWlv, 512, 512};
    pk[np++] = {DEC + 2, dWe, 1536, 512}; pk[np++] = {DEC + 0, dW1, 512, 512};
    pk[np++] = {DEC + 1, dW2, 512, 256};
    for (int i = 0; i < np; ++i)
        launch_pack(stream, P(pk[i].idx), pk[i].dst, pk[i].K, pk[i].N);
    {
        int n = B_ * S_ * 512;
        f2bf_kernel<<<(n + 255) / 256, 256, 0, stream>>>(enc_out, ench, n);
    }

    // --- encoder K/V (hoisted out of the scan, as in the reference) ---
    for (int l = 0; l < 2; ++l) {
        launch_gemm(stream, ench, 512, wk2[l], nullptr, ke[l], 512, B_ * S_, 512, 512, 0, true);
        launch_gemm(stream, ench, 512, wv2[l], nullptr, ve[l], 512, B_ * S_, 512, 512, 0, true);
    }

    init_kernel<<<50, 256, 0, stream>>>(trg_seq, (float*)d_out, prev);

    // --- autoregressive loop ---
    for (int t = 0; t < T_; ++t) {
        build_x_kernel<<<100, 256, 0, stream>>>(prev, x0, x0h, t);
        const float* linf = x0;
        const bf16t* linh = x0h;
        for (int l = 0; l < 2; ++l) {
            const int base = LB[l];
            // self-attn: incremental KV append (equal to the reference's
            // full-cache recompute under the causal mask: zero rows are masked)
            launch_gemm(stream, linh, 512, wq[l], nullptr, qb, 512, B_, 512, 512, 0, false);
            launch_gemm(stream, linh, 512, wk[l], nullptr, kc[l] + (size_t)t * 512, T_ * 512, B_, 512, 512, 0, true);
            launch_gemm(stream, linh, 512, wv[l], nullptr, vc[l] + (size_t)t * 512, T_ * 512, B_, 512, 512, 0, true);
            attn_kernel<<<B_ * H_, 256, 0, stream>>>(qb, kc[l], vc[l], aobh, t + 1, T_ * 512);
            launch_gemm(stream, aobh, 512, wo[l], P(base + 15), sab, 512, B_, 512, 512, 0, false);
            ln512_kernel<<<B_, 512, 0, stream>>>(sab, linf, P(base + 19), P(base + 12), hb, hbh, 0);
            // cross-attn
            launch_gemm(stream, hbh, 512, wq2[l], nullptr, qb, 512, B_, 512, 512, 0, false);
            attn_kernel<<<B_ * H_, 256, 0, stream>>>(qb, ke[l], ve[l], aobh, S_, S_ * 512);
            launch_gemm(stream, aobh, 512, wo2[l], P(base + 16), sab, 512, B_, 512, 512, 0, false);
            ln512_kernel<<<B_, 512, 0, stream>>>(sab, hb, P(base + 17), P(base + 10), h2b, h2bh, 0);
            // FFN
            launch_gemm(stream, h2bh, 512, wf1[l], P(base + 13), ffbh, 2048, B_, 2048, 512, 1, true);
            launch_gemm(stream, ffbh, 2048, wf2[l], P(base + 14), sab, 512, B_, 512, 2048, 0, false);
            ln512_kernel<<<B_, 512, 0, stream>>>(sab, h2b, P(base + 18), P(base + 11), xb, xbh, 0);
            linf = xb; linh = xbh;
        }
        // VAE heads
        build_headin_kernel<<<(B_ * 1280 + 255) / 256, 256, 0, stream>>>(x0, xb, inputemb, posinh, priinh, t);
        // posterior
        launch_gemm(stream, posinh, 1280, pWe, P(POS + 6), tmp1, 512, B_, 512, 1280, 0, false);
        ln512_kernel<<<B_, 512, 0, stream>>>(tmp1, nullptr, P(POS + 9), P(POS + 5), nullptr, hph, 2);
        launch_gemm(stream, hph, 512, pW1, P(POS + 4), hidbh, 512, B_, 512, 512, 2, true);
        launch_gemm(stream, hidbh, 512, pWmu, P(POS + 8), mu1b, 512, B_, 512, 512, 0, false);
        launch_gemm(stream, hidbh, 512, pWlv, P(POS + 7), lv1b, 512, B_, 512, 512, 0, false);
        z_kernel<<<100, 256, 0, stream>>>(mu1b, lv1b, zb, t);
        // prior
        launch_gemm(stream, priinh, 1024, rWe, P(PRI + 6), tmp1, 512, B_, 512, 1024, 0, false);
        ln512_kernel<<<B_, 512, 0, stream>>>(tmp1, nullptr, P(PRI + 9), P(PRI + 5), nullptr, hph, 2);
        launch_gemm(stream, hph, 512, rW1, P(PRI + 4), hidbh, 512, B_, 512, 512, 2, true);
        launch_gemm(stream, hidbh, 512, rWmu, P(PRI + 8), mu2b, 512, B_, 512, 512, 0, false);
        launch_gemm(stream, hidbh, 512, rWlv, P(PRI + 7), lv2b, 512, B_, 512, 512, 0, false);
        kld_kernel<<<100, 256, 0, stream>>>(mu1b, lv1b, mu2b, lv2b, (float*)d_out + KLD_OFF, t);
        // text-VAE decoder
        build_decin_kernel<<<(B_ * 1536 + 255) / 256, 256, 0, stream>>>(x0, xb, zb, decinh);
        launch_gemm(stream, decinh, 1536, dWe, P(DEC + 7), tmp1, 512, B_, 512, 1536, 0, false);
        ln512_kernel<<<B_, 512, 0, stream>>>(tmp1, nullptr, P(DEC + 9), P(DEC + 6), nullptr, hph, 2);
        launch_gemm(stream, hph, 512, dW1, P(DEC + 3), tmp1, 512, B_, 512, 512, 0, false);
        ln512_kernel<<<B_, 512, 0, stream>>>(tmp1, nullptr, P(DEC + 8), P(DEC + 5), nullptr, hd2h, 2);
        launch_gemm(stream, hd2h, 512, dW2, P(DEC + 4), doutb, 256, B_, 256, 512, 0, false);
        fake_kernel<<<50, 256, 0, stream>>>(doutb, (float*)d_out, prev, t);
    }
}
